// Deformer_46566035423580
// MI455X (gfx1250) — matrix operand order repaired and checked by reading, not compiled, not run on hardware
//
#include <hip/hip_runtime.h>
#include <hip/hip_bf16.h>
#include <math.h>

typedef __bf16 bf16;
typedef __attribute__((ext_vector_type(16))) __bf16 v16bf;
typedef __attribute__((ext_vector_type(8)))  __bf16 v8bf;
typedef __attribute__((ext_vector_type(8)))  float  v8f;
typedef __attribute__((ext_vector_type(4)))  int    i32x4;

#define B_   4
#define T_   4096
#define D_   2048
#define H_   16
#define DH_  128
#define BT_  (B_*T_)      // 16384 rows

#define MT   128          // M tile per workgroup
#define NT   128          // N tile per workgroup (== Dh, aligns to one head)
#define KS   32           // K step per WMMA
#define LDA  40           // padded LDS row stride (elements) -> conflict-free b128 reads

#if defined(__gfx1250__) && __has_builtin(__builtin_amdgcn_global_load_async_to_lds_b128)
#define ASYNC_LDS 1
#else
#define ASYNC_LDS 0
#endif

#if ASYNC_LDS
typedef __attribute__((address_space(1))) i32x4* gvec_p;
typedef __attribute__((address_space(3))) i32x4* lvec_p;
__device__ __forceinline__ void async_cp16(const bf16* g, bf16* l) {
  __builtin_amdgcn_global_load_async_to_lds_b128((gvec_p)g, (lvec_p)l, 0, 0);
}
__device__ __forceinline__ void wait_async0() {
#if __has_builtin(__builtin_amdgcn_s_wait_asynccnt)
  __builtin_amdgcn_s_wait_asynccnt(0);
#else
  asm volatile("s_wait_asynccnt 0x0" ::: "memory");
#endif
}
#endif

// ---------------- fp32 -> bf16 conversion (8 elems / thread) ----------------
__global__ __launch_bounds__(256)
void cvt_f32_bf16(const float* __restrict__ in, bf16* __restrict__ out, long long n) {
  long long i = ((long long)blockIdx.x * blockDim.x + threadIdx.x) * 8;
  if (i + 8 <= n) {
    float4 a = *(const float4*)(in + i);
    float4 b = *(const float4*)(in + i + 4);
    v8bf o;
    o[0] = (bf16)a.x; o[1] = (bf16)a.y; o[2] = (bf16)a.z; o[3] = (bf16)a.w;
    o[4] = (bf16)b.x; o[5] = (bf16)b.y; o[6] = (bf16)b.z; o[7] = (bf16)b.w;
    *(v8bf*)(out + i) = o;
  }
}

// ---------------- fused GEMM (Y = X @ W^T) + epilogue ----------------
// mode 0: RMS-norm over the 128-col head slice -> Out = normalized q/k
// mode 1: softplus + pos = clamp(t - s, 0, t)   -> Out = interp position
__global__ __launch_bounds__(256)
void gemm_proj(const bf16* __restrict__ Xb, const bf16* __restrict__ Wb,
               float* __restrict__ Out, int mode) {
  __shared__ bf16 As[2][MT * LDA];
  __shared__ bf16 Bs[2][NT * LDA];

  const int tid  = threadIdx.x;
  const int lane = tid & 31;
  const int wave = tid >> 5;        // 0..7 -> 16-row strip
  const int half = lane >> 4;       // 0/1
  const int l16  = lane & 15;

  const int mBlock = blockIdx.x * MT;
  const int nBlock = blockIdx.y * NT;

  // LDS fill mapping: 256 threads cover 128 rows x 2 k-halves (16 elems each)
  const int lr = tid >> 1;
  const int lk = (tid & 1) * 16;

  const int am = wave * 16 + l16;   // A-fragment row within block

  v8f zero = {};
  v8f acc[8];
  #pragma unroll
  for (int j = 0; j < 8; ++j) acc[j] = zero;

  union Frag { v16bf v; v8bf h[2]; };

#if ASYNC_LDS
  // ---- async global->LDS, double-buffered, one barrier per K-step ----
  auto issue = [&](int k0, int buf) {
    const bf16* ga = Xb + (size_t)(mBlock + lr) * D_ + k0 + lk;
    bf16*       la = &As[buf][lr * LDA + lk];
    async_cp16(ga,     la);
    async_cp16(ga + 8, la + 8);
    const bf16* gb = Wb + (size_t)(nBlock + lr) * D_ + k0 + lk;
    bf16*       lb = &Bs[buf][lr * LDA + lk];
    async_cp16(gb,     lb);
    async_cp16(gb + 8, lb + 8);
  };

  int cur = 0;
  issue(0, 0);
  for (int k0 = 0; k0 < D_; k0 += KS) {
    wait_async0();        // my writes into buf[cur] have landed in LDS
    __syncthreads();      // everyone's writes visible; everyone done reading buf[cur^1]
    if (k0 + KS < D_) issue(k0 + KS, cur ^ 1);   // prefetch overlaps compute

    const bf16* as = &As[cur][0];
    const bf16* bs = &Bs[cur][0];
    Frag af;
    af.h[0] = *(const v8bf*)&as[am * LDA + half * 8];
    af.h[1] = *(const v8bf*)&as[am * LDA + 16 + half * 8];
    Frag bfr[8];
    #pragma unroll
    for (int j = 0; j < 8; ++j) {
      const int bn = j * 16 + l16;
      bfr[j].h[0] = *(const v8bf*)&bs[bn * LDA + half * 8];
      bfr[j].h[1] = *(const v8bf*)&bs[bn * LDA + 16 + half * 8];
    }
    #pragma unroll
    for (int j = 0; j < 8; ++j)
      acc[j] = __builtin_amdgcn_wmma_f32_16x16x32_bf16(
          false, af.v, false, bfr[j].v, (short)0, acc[j], false, false);
    cur ^= 1;
  }
#else
  // ---- fallback: staged through VGPRs, two barriers per K-step ----
  for (int k0 = 0; k0 < D_; k0 += KS) {
    const bf16* ga = Xb + (size_t)(mBlock + lr) * D_ + k0 + lk;
    *(v8bf*)&As[0][lr * LDA + lk]     = *(const v8bf*)(ga);
    *(v8bf*)&As[0][lr * LDA + lk + 8] = *(const v8bf*)(ga + 8);
    const bf16* gb = Wb + (size_t)(nBlock + lr) * D_ + k0 + lk;
    *(v8bf*)&Bs[0][lr * LDA + lk]     = *(const v8bf*)(gb);
    *(v8bf*)&Bs[0][lr * LDA + lk + 8] = *(const v8bf*)(gb + 8);
    __syncthreads();

    const bf16* as = &As[0][0];
    const bf16* bs = &Bs[0][0];
    Frag af;
    af.h[0] = *(const v8bf*)&as[am * LDA + half * 8];
    af.h[1] = *(const v8bf*)&as[am * LDA + 16 + half * 8];
    Frag bfr[8];
    #pragma unroll
    for (int j = 0; j < 8; ++j) {
      const int bn = j * 16 + l16;
      bfr[j].h[0] = *(const v8bf*)&bs[bn * LDA + half * 8];
      bfr[j].h[1] = *(const v8bf*)&bs[bn * LDA + 16 + half * 8];
    }
    #pragma unroll
    for (int j = 0; j < 8; ++j)
      acc[j] = __builtin_amdgcn_wmma_f32_16x16x32_bf16(
          false, af.v, false, bfr[j].v, (short)0, acc[j], false, false);
    __syncthreads();
  }
#endif

  // Epilogue. C/D layout: acc[j][r] = element (row = wave*16 + half*8 + r,
  // col = nBlock + j*16 + l16). A row's 128 cols live in 8 j-tiles x 16 lanes
  // of this half-wave -> shfl_xor(1,2,4,8) reduces within the 16-lane group.
  #pragma unroll
  for (int r = 0; r < 8; ++r) {
    const int rowG = mBlock + wave * 16 + half * 8 + r;
    if (mode == 0) {
      float ss = 0.f;
      #pragma unroll
      for (int j = 0; j < 8; ++j) { float v = acc[j][r]; ss += v * v; }
      ss += __shfl_xor(ss, 1, 32);
      ss += __shfl_xor(ss, 2, 32);
      ss += __shfl_xor(ss, 4, 32);
      ss += __shfl_xor(ss, 8, 32);
      const float scale = rsqrtf(ss * (1.0f / 128.0f) + 1.1920929e-7f);
      #pragma unroll
      for (int j = 0; j < 8; ++j)
        Out[(size_t)rowG * D_ + nBlock + j * 16 + l16] = acc[j][r] * scale;
    } else {
      const float tf = (float)(rowG & (T_ - 1));   // t index within sequence
      #pragma unroll
      for (int j = 0; j < 8; ++j) {
        const float v  = acc[j][r];
        const float sp = fmaxf(v, 0.f) + log1pf(expf(-fabsf(v)));  // stable softplus
        const float p  = fminf(fmaxf(tf - sp, 0.f), tf);
        Out[(size_t)rowG * D_ + nBlock + j * 16 + l16] = p;
      }
    }
  }
}

// ---------------- gather + linear interp + product ----------------
__global__ __launch_bounds__(256)
void interp_mul(const float* __restrict__ qn, const float* __restrict__ kn,
                const float* __restrict__ pq, const float* __restrict__ pk,
                float* __restrict__ out) {
  const size_t idx = (size_t)blockIdx.x * blockDim.x + threadIdx.x;
  const size_t TD  = (size_t)T_ * D_;
  const size_t b   = idx / TD;
  const size_t rem = idx - b * TD;
  const int    c   = (int)(rem % D_);
  const size_t base = b * TD + c;

  const float posq = pq[idx];
  const float fq   = floorf(posq);
  int q0 = (int)fq; q0 = max(0, min(q0, T_ - 1));
  const int q1 = min(q0 + 1, T_ - 1);
  const float frq = posq - fq;
  const float a0 = qn[base + (size_t)q0 * D_];
  const float a1 = qn[base + (size_t)q1 * D_];
  const float qd = a0 + (a1 - a0) * frq;

  const float posk = pk[idx];
  const float fk   = floorf(posk);
  int k0 = (int)fk; k0 = max(0, min(k0, T_ - 1));
  const int k1 = min(k0 + 1, T_ - 1);
  const float frk = posk - fk;
  const float b0 = kn[base + (size_t)k0 * D_];
  const float b1 = kn[base + (size_t)k1 * D_];
  const float kd = b0 + (b1 - b0) * frk;

  out[idx] = qd * kd;
}

extern "C" void kernel_launch(void* const* d_in, const int* in_sizes, int n_in,
                              void* d_out, int out_size, void* d_ws, size_t ws_size,
                              hipStream_t stream) {
  const float* x   = (const float*)d_in[0];
  const float* Wq  = (const float*)d_in[1];
  const float* Wk  = (const float*)d_in[2];
  const float* Wsq = (const float*)d_in[3];
  const float* Wsk = (const float*)d_in[4];
  float* out = (float*)d_out;

  char* ws = (char*)d_ws;
  const size_t nX = (size_t)BT_ * D_;   // 33.55M elems
  const size_t nW = (size_t)D_ * D_;    // 4.19M elems

  bf16* xb  = (bf16*)ws;  ws += nX * sizeof(bf16);
  bf16* wqb = (bf16*)ws;  ws += nW * sizeof(bf16);
  bf16* wkb = (bf16*)ws;  ws += nW * sizeof(bf16);
  bf16* wsq = (bf16*)ws;  ws += nW * sizeof(bf16);
  bf16* wsk = (bf16*)ws;  ws += nW * sizeof(bf16);
  float* qn   = (float*)ws; ws += nX * sizeof(float);
  float* kn   = (float*)ws; ws += nX * sizeof(float);
  float* posq = (float*)ws; ws += nX * sizeof(float);
  float* posk = (float*)ws; ws += nX * sizeof(float);

  // 1) convert operands to bf16 once (halves GEMM traffic; x fits in L2 as bf16)
  cvt_f32_bf16<<<(unsigned)(nX / 8 / 256), 256, 0, stream>>>(x,   xb,  (long long)nX);
  cvt_f32_bf16<<<(unsigned)(nW / 8 / 256), 256, 0, stream>>>(Wq,  wqb, (long long)nW);
  cvt_f32_bf16<<<(unsigned)(nW / 8 / 256), 256, 0, stream>>>(Wk,  wkb, (long long)nW);
  cvt_f32_bf16<<<(unsigned)(nW / 8 / 256), 256, 0, stream>>>(Wsq, wsq, (long long)nW);
  cvt_f32_bf16<<<(unsigned)(nW / 8 / 256), 256, 0, stream>>>(Wsk, wsk, (long long)nW);

  // 2) four fused WMMA GEMMs with in-register epilogues
  dim3 grid(BT_ / MT, D_ / NT);   // 128 x 16 workgroups
  gemm_proj<<<grid, 256, 0, stream>>>(xb, wqb, qn,   0);
  gemm_proj<<<grid, 256, 0, stream>>>(xb, wkb, kn,   0);
  gemm_proj<<<grid, 256, 0, stream>>>(xb, wsq, posq, 1);
  gemm_proj<<<grid, 256, 0, stream>>>(xb, wsk, posk, 1);

  // 3) deformable gather + lerp + product
  interp_mul<<<(unsigned)(nX / 256), 256, 0, stream>>>(qn, kn, posq, posk, out);
}
